// InteractionBlock_13005160972695
// MI455X (gfx1250) — matrix operand order repaired and checked by reading, not compiled, not run on hardware
//
#include <hip/hip_runtime.h>

// ---------------------------------------------------------------------------
// MI455X (gfx1250) implementation of the DimeNet-style InteractionBlock.
//
// Dominant op: bilinear contraction recast as [A=1e6, K=1024] x [1024, 128]
// GEMM with on-the-fly rank-1 A rows; all GEMMs on v_wmma_f32_16x16x32_bf16
// (f32 accumulate). Weights pre-swizzled to WMMA B-fragment order. The main
// kernel stages B fragments through LDS with the Tensor Data Mover
// (tensor_load_to_lds + s_wait_tensorcnt), double buffered.
// ---------------------------------------------------------------------------

typedef __attribute__((ext_vector_type(16))) __bf16   v16bf;
typedef __attribute__((ext_vector_type(8)))  float    v8f;
typedef __attribute__((ext_vector_type(4)))  unsigned u32x4;
typedef __attribute__((ext_vector_type(8)))  unsigned u32x8;
typedef __attribute__((ext_vector_type(2)))  float    f32x2;

union Frag {            // 32 bytes = one wave-lane's share of a 16x32 bf16 tile
  u32x4    q[2];
  unsigned u[8];
  v16bf    v;
};

// ---- scalar helpers -------------------------------------------------------

static __device__ __forceinline__ float bitsf(unsigned u) {
  float f;
  __builtin_memcpy(&f, &u, 4);
  return f;
}
static __device__ __forceinline__ unsigned fbits(float f) {
  unsigned u;
  __builtin_memcpy(&u, &f, 4);
  return u;
}
// Full RNE conversion (cold path: one-time weight conversion).
static __device__ __forceinline__ unsigned short f2bf_rne(float f) {
  unsigned u = fbits(f);
  unsigned r = u + 0x7FFFu + ((u >> 16) & 1u);
  return (unsigned short)(r >> 16);
}
static __device__ __forceinline__ unsigned pack2_rne(float a, float b) {
  return (unsigned)f2bf_rne(a) | ((unsigned)f2bf_rne(b) << 16);
}
// Fast round-half-up conversions (hot paths; f32 accumulate downstream).
static __device__ __forceinline__ unsigned short f2bfr(float f) {
  return (unsigned short)((fbits(f) + 0x8000u) >> 16);
}
// pack two floats' (rounded) high halves into one dword with v_perm_b32
static __device__ __forceinline__ unsigned pack2r(float a, float b) {
  return __builtin_amdgcn_perm(fbits(b) + 0x8000u, fbits(a) + 0x8000u,
                               0x07060302u);
}
// packed bf16 * bf16 (as f32 math) -> packed bf16
static __device__ __forceinline__ unsigned mulpk(unsigned a, unsigned b) {
  float al = bitsf(a << 16), ah = bitsf(a & 0xffff0000u);
  float bl = bitsf(b << 16), bh = bitsf(b & 0xffff0000u);
  return pack2r(al * bl, ah * bh);
}
static __device__ __forceinline__ float siluf(float x) {
  return x * __builtin_amdgcn_rcpf(1.0f + __expf(-x));
}
static __device__ __forceinline__ v8f wmma_bf16(const Frag& A, const Frag& B, v8f c) {
  return __builtin_amdgcn_wmma_f32_16x16x32_bf16(false, A.v, false, B.v,
                                                 (short)0, c, false, false);
}
static __device__ __forceinline__ void zero_acc(v8f acc[8]) {
#pragma unroll
  for (int nt = 0; nt < 8; nt++)
#pragma unroll
    for (int v = 0; v < 8; v++) acc[nt][v] = 0.0f;
}

// ---- Tensor Data Mover: async copy of 8KB (2048 dwords) global -> LDS -----
// D# groups per CDNA5 ISA ch.8: group0 {count/lds_addr/global_addr/type},
// group1 {data_size=4B, tensor_dim0=2048, tile_dim0=2048, dims1..=1},
// groups 2/3 present (dims 2..4 = 1) so all four VADDR operands are real.
static __device__ __forceinline__ void tdm_copy_8k(const unsigned* gsrc,
                                                   unsigned lds_off) {
  unsigned long long ga = (unsigned long long)(uintptr_t)gsrc;
  u32x4 g0;
  g0.x = 1u;                                   // count=1, user descriptor
  g0.y = lds_off;                              // LDS byte address
  g0.z = (unsigned)ga;                         // global_addr[31:0]
  g0.w = (unsigned)((ga >> 32) & 0x01FFFFFFu)  // global_addr[56:32]
         | (2u << 30);                         // type = 2 ("image")
  u32x8 g1;
  g1.s0 = 0x00020000u;                         // data_size = 2 (4 bytes)
  g1.s1 = 2048u << 16;                         // tensor_dim0[15:0] @ bits 63:48
  g1.s2 = 1u << 16;                            // tensor_dim1 = 1
  g1.s3 = 2048u << 16;                         // tile_dim0 = 2048
  g1.s4 = 1u;                                  // tile_dim1 = 1, tile_dim2 = 0
  g1.s5 = 2048u;                               // tensor_dim0_stride = 2048
  g1.s6 = 0u;
  g1.s7 = 0u;
  u32x4 g2;
  g2.x = 1u;                                   // tensor_dim2 = 1
  g2.y = 1u;                                   // tensor_dim3 = 1
  g2.z = 0u;                                   // tensor_dim2_stride = 0
  g2.w = 0u;                                   // tile_dim3 = 0 (unused)
  u32x4 g3;
  g3.x = 0u;                                   // tensor_dim3_stride lo
  g3.y = 1u << 16;                             // tensor_dim4 = 1
  g3.z = 0u;                                   // tile_dim4 = 0 (unused)
  g3.w = 0u;
  asm volatile("tensor_load_to_lds %0, %1, %2, %3"
               :: "s"(g0), "s"(g1), "s"(g2), "s"(g3)
               : "memory");
}

// ---- GEMM helpers ---------------------------------------------------------
// A fragment layout (16-bit A, 16x32, CDNA5 ISA 7.12.2):
//   lane l: row m = l&15, half h = l>>4
//   vgpr v<4 : K = 8h + 2v, +1        vgpr v>=4 : K = 16 + 8h + 2(v-4), +1
// B fragments pre-swizzled in memory: frag f = kt*8+nt holds 256 dwords
// (lane*8+v), dword v packing (K = kt*32 + 8h + 2(v&3) + 16(v>>2), K+1) at col nt*16+(l&15).

static __device__ __forceinline__ void gemm_rowf32(const float* __restrict__ Abase,
                                                   const unsigned* __restrict__ Wf,
                                                   int lane, v8f acc[8]) {
  const int h = lane >> 4;
  const f32x2* p = (const f32x2*)(Abase + (size_t)(lane & 15) * 128);
#pragma unroll
  for (int kc = 0; kc < 4; kc++) {
    Frag Afr;
#pragma unroll
    for (int v = 0; v < 4; v++) {
      f32x2 f0 = p[kc * 16 + 4 * h + v];
      f32x2 f1 = p[kc * 16 + 8 + 4 * h + v];
      Afr.u[v]     = pack2r(f0.x, f0.y);
      Afr.u[v + 4] = pack2r(f1.x, f1.y);
    }
#pragma unroll
    for (int nt = 0; nt < 8; nt++) {
      Frag B;
      const u32x4* bp = (const u32x4*)(Wf + ((size_t)(kc * 8 + nt) * 256 + lane * 8));
      B.q[0] = bp[0];
      B.q[1] = bp[1];
      acc[nt] = wmma_bf16(Afr, B, acc[nt]);
    }
  }
}

static __device__ __forceinline__ void gemm_ldsA(const unsigned short* __restrict__ act,
                                                 const unsigned* __restrict__ Wf,
                                                 int lane, v8f acc[8]) {
  const int h = lane >> 4;
  const u32x4* xr = (const u32x4*)(act + (size_t)(lane & 15) * 128);
#pragma unroll
  for (int kc = 0; kc < 4; kc++) {
    Frag Afr;
    Afr.q[0] = xr[kc * 4 + h];
    Afr.q[1] = xr[kc * 4 + 2 + h];
#pragma unroll
    for (int nt = 0; nt < 8; nt++) {
      Frag B;
      const u32x4* bp = (const u32x4*)(Wf + ((size_t)(kc * 8 + nt) * 256 + lane * 8));
      B.q[0] = bp[0];
      B.q[1] = bp[1];
      acc[nt] = wmma_bf16(Afr, B, acc[nt]);
    }
  }
}

// ---------------------------------------------------------------------------
// Weight conversion: f32 [K x 128] (mode 0) or w_bil[i][j][l] (mode 1) ->
// bf16 WMMA B-fragment order. One thread per packed dword. RNE (cold path).
// ---------------------------------------------------------------------------
__global__ void k_frag_convert(const float* __restrict__ src,
                               unsigned* __restrict__ dst, int KT, int mode) {
  int tid  = blockIdx.x * 256 + threadIdx.x;
  int v    = tid & 7;
  int lane = (tid >> 3) & 31;
  int frag = tid >> 8;
  if (frag >= KT * 8) return;
  int nt = frag & 7, kt = frag >> 3;
  int h = lane >> 4, nl = lane & 15;
  int k = kt * 32 + 8 * h + 2 * (v & 3) + 16 * (v >> 2);
  int n = nt * 16 + nl;
  float f0, f1;
  if (mode == 0) {
    f0 = src[(size_t)k * 128 + n];
    f1 = src[(size_t)(k + 1) * 128 + n];
  } else {  // w_bil[i=n][j][l] flat = n*1024 + k
    f0 = src[(size_t)n * 1024 + k];
    f1 = src[(size_t)n * 1024 + k + 1];
  }
  dst[(size_t)frag * 256 + lane * 8 + v] = pack2_rne(f0, f1);
}

__global__ void k_zero4(u32x4* __restrict__ p, size_t n4) {
  size_t i = (size_t)blockIdx.x * blockDim.x + threadIdx.x;
  if (i < n4) p[i] = (u32x4)0u;
}

// ---------------------------------------------------------------------------
// Per-edge precompute: T = silu(m_ji @ W_mkj + b) (WMMA), U = e_rbf @ W_e.
// ---------------------------------------------------------------------------
__global__ void __launch_bounds__(256)
k_edge_pre(const float* __restrict__ m_ji, const float* __restrict__ e_rbf,
           const float* __restrict__ W_e, const float* __restrict__ b_mkj,
           const unsigned* __restrict__ Wf_mkj,
           unsigned short* __restrict__ T, unsigned short* __restrict__ U, int E) {
  __shared__ float We_s[6 * 128];
  int tid = threadIdx.x;
  for (int i = tid; i < 6 * 128; i += 256) We_s[i] = W_e[i];
  __syncthreads();

  int wv = tid >> 5, lane = tid & 31;
  int tiles = (E + 15) / 16;
  int tile = blockIdx.x * 8 + wv;
  if (tile >= tiles) return;
  int e0 = tile * 16;
  int nl = lane & 15, h = lane >> 4;

  v8f acc[8];
  zero_acc(acc);
  gemm_rowf32(m_ji + (size_t)e0 * 128, Wf_mkj, lane, acc);
#pragma unroll
  for (int nt = 0; nt < 8; nt++) {
    int n = nt * 16 + nl;
    float b = b_mkj[n];
#pragma unroll
    for (int v = 0; v < 8; v++) {
      int m = v + 8 * h;
      T[(size_t)(e0 + m) * 128 + n] = f2bfr(siluf(acc[nt][v] + b));
    }
  }

  {  // U = e_rbf @ W_e (K=6)
    int m = nl;
    const float* er = e_rbf + (size_t)(e0 + m) * 6;
    float r0 = er[0], r1 = er[1], r2 = er[2], r3 = er[3], r4 = er[4], r5 = er[5];
#pragma unroll 8
    for (int c = 0; c < 64; c++) {
      int n = 64 * h + c;
      float s = r0 * We_s[n] + r1 * We_s[128 + n] + r2 * We_s[256 + n] +
                r3 * We_s[384 + n] + r4 * We_s[512 + n] + r5 * We_s[640 + n];
      U[(size_t)(e0 + m) * 128 + n] = f2bfr(s);
    }
  }
}

// ---------------------------------------------------------------------------
// Main per-angle bilinear GEMM + segment-sum.
// B fragments (8KB per K-step) staged via TDM, double-buffered in LDS.
// ---------------------------------------------------------------------------
__global__ void __launch_bounds__(256)
k_angle(const float* __restrict__ a_sbf, const int* __restrict__ kj_idx,
        const int* __restrict__ ji_idx, const float* __restrict__ W_a,
        const unsigned short* __restrict__ T, const unsigned short* __restrict__ U,
        const unsigned* __restrict__ WBf, float* __restrict__ directed, int Anum) {
  __shared__ float Wa_s[49 * 8];
  __shared__ __attribute__((aligned(16))) unsigned short x_s[8][16 * 128];
  __shared__ float a_s[8][16 * 8];
  __shared__ int kj_s[8][16];
  __shared__ int ji_s[8][16];
  __shared__ __attribute__((aligned(16))) unsigned bst[2][8 * 256];

  int tid = threadIdx.x, wv = tid >> 5, lane = tid & 31;
  for (int i = tid; i < 49 * 8; i += 256) Wa_s[i] = W_a[i];

  int tiles = (Anum + 15) / 16;
  int tile = blockIdx.x * 8 + wv;
  bool valid = tile < tiles;
  if (!valid) tile = tiles - 1;
  int a0 = tile * 16;
  int nl = lane & 15, h = lane >> 4;

  if (lane < 16) {
    int w = a0 + lane;
    if (w >= Anum) w = Anum - 1;
    kj_s[wv][lane] = kj_idx[w];
    ji_s[wv][lane] = ji_idx[w];
  }
  // kick off the first B-fragment tile DMA early (wave 0 only)
  if (wv == 0) tdm_copy_8k(WBf, (unsigned)(uintptr_t)&bst[0][0]);
  __syncthreads();  // Wa_s + kj/ji visible

  if (lane < 16) {  // a[m,:] = a_sbf[m] @ W_a
    int w = a0 + lane;
    if (w >= Anum) w = Anum - 1;
    const float* sr = a_sbf + (size_t)w * 49;
#pragma unroll
    for (int j = 0; j < 8; j++) {
      float s = 0.0f;
      for (int i = 0; i < 49; i++) s += sr[i] * Wa_s[i * 8 + j];
      a_s[wv][lane * 8 + j] = s;
    }
  }

  {  // x[m,:] = T[kj[m],:] * U[ji[m],:]
    int m = nl;
    int kj = kj_s[wv][m], ji = ji_s[wv][m];
    const u32x4* tp = (const u32x4*)(T + (size_t)kj * 128 + 64 * h);
    const u32x4* up = (const u32x4*)(U + (size_t)ji * 128 + 64 * h);
    u32x4* xp = (u32x4*)(&x_s[wv][m * 128 + 64 * h]);
#pragma unroll
    for (int c = 0; c < 8; c++) {
      u32x4 t = tp[c], u = up[c], o;
      o.x = mulpk(t.x, u.x);
      o.y = mulpk(t.y, u.y);
      o.z = mulpk(t.z, u.z);
      o.w = mulpk(t.w, u.w);
      xp[c] = o;
    }
  }
  __builtin_amdgcn_s_wait_tensorcnt(0);  // buf0 DMA complete (wave 0)
  __syncthreads();                       // a_s, x_s, bst[0] visible

  v8f acc[8];
  zero_acc(acc);

  for (int kc = 0; kc < 32; kc++) {
    int buf = kc & 1;
    // overlap: DMA the next 8KB of B fragments while we compute this step
    if (kc + 1 < 32 && wv == 0)
      tdm_copy_8k(WBf + (size_t)(kc + 1) * 2048,
                  (unsigned)(uintptr_t)&bst[buf ^ 1][0]);

    // scaled A fragment: z = a[m, kc>>2] * x[m, (kc&3)*32 ..]
    float aj = a_s[wv][nl * 8 + (kc >> 2)];
    const u32x4* xr = (const u32x4*)(&x_s[wv][nl * 128]);
    int lc = kc & 3;
    Frag X, Az;
    X.q[0] = xr[lc * 4 + h];
    X.q[1] = xr[lc * 4 + 2 + h];
#pragma unroll
    for (int v = 0; v < 8; v++) {
      unsigned xv = X.u[v];
      Az.u[v] = pack2r(aj * bitsf(xv << 16), aj * bitsf(xv & 0xffff0000u));
    }
#pragma unroll
    for (int nt = 0; nt < 8; nt++) {
      Frag B;
      const u32x4* bp = (const u32x4*)(&bst[buf][nt * 256 + lane * 8]);
      B.q[0] = bp[0];
      B.q[1] = bp[1];
      acc[nt] = wmma_bf16(Az, B, acc[nt]);
    }
    __builtin_amdgcn_s_wait_tensorcnt(0);  // next buffer landed (wave 0)
    __syncthreads();                       // everyone done with bst[buf]
  }

  if (valid) {  // segment_sum via f32 atomics
#pragma unroll
    for (int nt = 0; nt < 8; nt++) {
      int n = nt * 16 + nl;
#pragma unroll
      for (int v = 0; v < 8; v++) {
        int m = v + 8 * h;
        atomicAdd(&directed[(size_t)ji_s[wv][m] * 128 + n], acc[nt][v]);
      }
    }
  }
}

// ---------------------------------------------------------------------------
// Per-edge epilogue: 8 chained [16,128]x[128,128] WMMA GEMMs; activations
// shuttle D-layout -> A-layout through per-wave LDS tiles.
// ---------------------------------------------------------------------------
__global__ void __launch_bounds__(256)
k_post(const float* __restrict__ m_ji, const float* __restrict__ directed,
       const float* __restrict__ b_mji, const float* __restrict__ b_post,
       const float* __restrict__ res_b, const unsigned* __restrict__ Wf_mji,
       const unsigned* __restrict__ Wf_post, const unsigned* __restrict__ Wf_res,
       float* __restrict__ out, int E) {
  __shared__ __attribute__((aligned(16))) unsigned short act[8][16 * 128];
  __shared__ float resid[8][16 * 128];

  int tid = threadIdx.x, wv = tid >> 5, lane = tid & 31;
  int tiles = (E + 15) / 16;
  int tile = blockIdx.x * 8 + wv;
  if (tile >= tiles) return;  // no block-wide barriers below
  int e0 = tile * 16;
  int nl = lane & 15, h = lane >> 4;
  unsigned short* A_act = &act[wv][0];
  float* A_res = &resid[wv][0];

  v8f acc[8];

  // cur = directed + silu(m_ji @ W_mji + b_mji)
  zero_acc(acc);
  gemm_rowf32(m_ji + (size_t)e0 * 128, Wf_mji, lane, acc);
#pragma unroll
  for (int nt = 0; nt < 8; nt++) {
    int n = nt * 16 + nl;
    float b = b_mji[n];
#pragma unroll
    for (int v = 0; v < 8; v++) {
      int m = v + 8 * h;
      float cur = directed[(size_t)(e0 + m) * 128 + n] + siluf(acc[nt][v] + b);
      A_res[m * 128 + n] = cur;
      A_act[m * 128 + n] = f2bfr(cur);
    }
  }
  __builtin_amdgcn_wave_barrier();

  // residual block 0, layer 0
  zero_acc(acc);
  gemm_ldsA(A_act, Wf_res + (size_t)0 * 8192, lane, acc);
#pragma unroll
  for (int nt = 0; nt < 8; nt++) {
    int n = nt * 16 + nl;
    float b = res_b[0 * 128 + n];
#pragma unroll
    for (int v = 0; v < 8; v++) {
      int m = v + 8 * h;
      A_act[m * 128 + n] = f2bfr(siluf(acc[nt][v] + b));
    }
  }
  __builtin_amdgcn_wave_barrier();

  // residual block 0, layer 1 + residual
  zero_acc(acc);
  gemm_ldsA(A_act, Wf_res + (size_t)1 * 8192, lane, acc);
  __builtin_amdgcn_wave_barrier();
#pragma unroll
  for (int nt = 0; nt < 8; nt++) {
    int n = nt * 16 + nl;
    float b = res_b[1 * 128 + n];
#pragma unroll
    for (int v = 0; v < 8; v++) {
      int m = v + 8 * h;
      float x = siluf(acc[nt][v] + b) + A_res[m * 128 + n];
      A_act[m * 128 + n] = f2bfr(x);
    }
  }
  __builtin_amdgcn_wave_barrier();

  // out2 = silu(r @ W_post + b_post) + m_ji
  zero_acc(acc);
  gemm_ldsA(A_act, Wf_post, lane, acc);
  __builtin_amdgcn_wave_barrier();
#pragma unroll
  for (int nt = 0; nt < 8; nt++) {
    int n = nt * 16 + nl;
    float b = b_post[n];
#pragma unroll
    for (int v = 0; v < 8; v++) {
      int m = v + 8 * h;
      float x = siluf(acc[nt][v] + b) + m_ji[(size_t)(e0 + m) * 128 + n];
      A_res[m * 128 + n] = x;
      A_act[m * 128 + n] = f2bfr(x);
    }
  }
  __builtin_amdgcn_wave_barrier();

  // residual blocks 1 and 2
#pragma unroll 1
  for (int k = 1; k < 3; k++) {
    zero_acc(acc);
    gemm_ldsA(A_act, Wf_res + (size_t)(2 * k) * 8192, lane, acc);
    __builtin_amdgcn_wave_barrier();
#pragma unroll
    for (int nt = 0; nt < 8; nt++) {
      int n = nt * 16 + nl;
      float b = res_b[(2 * k) * 128 + n];
#pragma unroll
      for (int v = 0; v < 8; v++) {
        int m = v + 8 * h;
        A_act[m * 128 + n] = f2bfr(siluf(acc[nt][v] + b));
      }
    }
    __builtin_amdgcn_wave_barrier();

    zero_acc(acc);
    gemm_ldsA(A_act, Wf_res + (size_t)(2 * k + 1) * 8192, lane, acc);
    __builtin_amdgcn_wave_barrier();
#pragma unroll
    for (int nt = 0; nt < 8; nt++) {
      int n = nt * 16 + nl;
      float b = res_b[(2 * k + 1) * 128 + n];
#pragma unroll
      for (int v = 0; v < 8; v++) {
        int m = v + 8 * h;
        float x = siluf(acc[nt][v] + b) + A_res[m * 128 + n];
        if (k == 2) {
          out[(size_t)(e0 + m) * 128 + n] = x;
        } else {
          A_res[m * 128 + n] = x;
          A_act[m * 128 + n] = f2bfr(x);
        }
      }
    }
    __builtin_amdgcn_wave_barrier();
  }
}

// ---------------------------------------------------------------------------
// Host launcher
// ---------------------------------------------------------------------------
extern "C" void kernel_launch(void* const* d_in, const int* in_sizes, int n_in,
                              void* d_out, int out_size, void* d_ws, size_t ws_size,
                              hipStream_t stream) {
  const float* m_ji   = (const float*)d_in[0];
  const float* e_rbf  = (const float*)d_in[1];
  const float* a_sbf  = (const float*)d_in[2];
  const int*   kj_idx = (const int*)d_in[3];
  const int*   ji_idx = (const int*)d_in[4];
  const float* W_mkj  = (const float*)d_in[5];
  const float* b_mkj  = (const float*)d_in[6];
  const float* W_e    = (const float*)d_in[7];
  const float* W_a    = (const float*)d_in[8];
  const float* w_bil  = (const float*)d_in[9];
  const float* W_mji  = (const float*)d_in[10];
  const float* b_mji  = (const float*)d_in[11];
  const float* W_post = (const float*)d_in[12];
  const float* b_post = (const float*)d_in[13];
  const float* res_W  = (const float*)d_in[14];
  const float* res_b  = (const float*)d_in[15];

  const int E = in_sizes[0] / 128;
  const int A = in_sizes[3];

  char* ws = (char*)d_ws;
  size_t off = 0;
  float* directed = (float*)(ws + off);            off += (size_t)E * 128 * 4;
  unsigned short* T = (unsigned short*)(ws + off); off += (size_t)E * 128 * 2;
  unsigned short* U = (unsigned short*)(ws + off); off += (size_t)E * 128 * 2;
  unsigned* WBf    = (unsigned*)(ws + off);        off += (size_t)1024 * 128 * 2;
  unsigned* Wf_mkj = (unsigned*)(ws + off);        off += 128 * 128 * 2;
  unsigned* Wf_mji = (unsigned*)(ws + off);        off += 128 * 128 * 2;
  unsigned* Wf_post= (unsigned*)(ws + off);        off += 128 * 128 * 2;
  unsigned* Wf_res = (unsigned*)(ws + off);        off += (size_t)6 * 128 * 128 * 2;
  if (ws_size < off) return;

  // 1) weight conversion / fragment swizzle (bf16)
  k_frag_convert<<<32 * 8, 256, 0, stream>>>(w_bil, WBf, 32, 1);
  k_frag_convert<<<4 * 8, 256, 0, stream>>>(W_mkj, Wf_mkj, 4, 0);
  k_frag_convert<<<4 * 8, 256, 0, stream>>>(W_mji, Wf_mji, 4, 0);
  k_frag_convert<<<4 * 8, 256, 0, stream>>>(W_post, Wf_post, 4, 0);
  for (int i = 0; i < 6; i++)
    k_frag_convert<<<4 * 8, 256, 0, stream>>>(res_W + (size_t)i * 16384,
                                              Wf_res + (size_t)i * 8192, 4, 0);

  // 2) zero the segment-sum target
  {
    size_t n4 = (size_t)E * 32;
    int blocks = (int)((n4 + 255) / 256);
    k_zero4<<<blocks, 256, 0, stream>>>((u32x4*)directed, n4);
  }

  // 3) per-edge precompute T, U
  int etiles = (E + 15) / 16;
  int eblocks = (etiles + 7) / 8;
  k_edge_pre<<<eblocks, 256, 0, stream>>>(m_ji, e_rbf, W_e, b_mkj, Wf_mkj, T, U, E);

  // 4) main bilinear GEMM + segment-sum (TDM-staged B)
  int atiles = (A + 15) / 16;
  int ablocks = (atiles + 7) / 8;
  k_angle<<<ablocks, 256, 0, stream>>>(a_sbf, kj_idx, ji_idx, W_a, T, U, WBf,
                                       directed, A);

  // 5) per-edge epilogue chain -> d_out
  k_post<<<eblocks, 256, 0, stream>>>(m_ji, directed, b_mji, b_post, res_b,
                                      Wf_mji, Wf_post, Wf_res, (float*)d_out, E);
}
